// OptimizedKANLayer_30812095381506
// MI455X (gfx1250) — matrix operand order repaired and checked by reading, not compiled, not run on hardware
//
#include <hip/hip_runtime.h>
#include <hip/hip_bf16.h>

// ---------------------------------------------------------------------------
// KAN layer for MI455X (gfx1250, wave32).
//
// Strategy: knots are a broadcast linspace across (OUT_DIM, IN_DIM) in the
// reference, so the basis is independent of the output channel. The layer is
// then exactly one GEMM:
//   Phi[2048 tokens x 768 feats]  (512 spline-basis feats + 256 activation feats)
//   Wm [64 outs   x 768]          (control_points ++ 0.1*activation_weights)
//   out[b,o,n] = sum_k Phi[b*512+n, k] * Wm[o, k]
// Prep runs in f32 (faithful Cox-de Boor with EPS and half-open intervals),
// the contraction runs on the matrix pipe via v_wmma_f32_16x16x32_f16 with
// f32 accumulation. Whole problem is L2-resident (<4MB) => latency bound;
// the WMMA path minimizes instruction count for the contraction.
// ---------------------------------------------------------------------------

#define IN_DIM      64
#define OUT_DIM     64
#define NUM_SPLINES 8
#define DEGREE      3
#define NKNOT       12            // NUM_SPLINES + DEGREE + 1
#define EPS_F       1e-8f
#define SEQ         512
#define NTOK        2048          // 4 * 512
#define KSPL        512           // IN_DIM * NUM_SPLINES
#define KACT        256           // IN_DIM * 4
#define KTOT        768           // KSPL + KACT

typedef _Float16 half16 __attribute__((ext_vector_type(16)));
typedef _Float16 half8  __attribute__((ext_vector_type(8)));
typedef _Float16 half4  __attribute__((ext_vector_type(4)));
typedef float    float8 __attribute__((ext_vector_type(8)));

// ---------------------------------------------------------------------------
// Kernel 1: build feature matrix Phi (f16), one thread per (token, d).
// ---------------------------------------------------------------------------
__global__ __launch_bounds__(256)
void kan_features(const float* __restrict__ x,
                  const float* __restrict__ knots,
                  _Float16* __restrict__ Phi) {
    int tid   = blockIdx.x * blockDim.x + threadIdx.x;  // NTOK*IN_DIM threads
    int token = tid >> 6;
    int d     = tid & 63;

    float xv = x[token * IN_DIM + d];

    // knots is broadcast across OUT_DIM in the reference; use the o=0 row.
    float t[NKNOT];
#pragma unroll
    for (int i = 0; i < NKNOT; ++i) t[i] = knots[d * NKNOT + i];

    // Degree-0 basis: half-open intervals, exactly as the reference.
    float B[NKNOT - 1];
#pragma unroll
    for (int i = 0; i < NKNOT - 1; ++i)
        B[i] = (xv >= t[i] && xv < t[i + 1]) ? 1.0f : 0.0f;

    // Cox-de Boor recursion with the reference's EPS regularization.
#pragma unroll
    for (int k = 1; k <= DEGREE; ++k) {
#pragma unroll
        for (int i = 0; i < NKNOT - 1 - k; ++i) {
            float c1 = (xv - t[i])         / (t[i + k]     - t[i]     + EPS_F);
            float c2 = (t[i + k + 1] - xv) / (t[i + k + 1] - t[i + 1] + EPS_F);
            B[i] = c1 * B[i] + c2 * B[i + 1];
        }
    }

    // Spline features: Phi[token][d*8 + s]
    half8 hb;
#pragma unroll
    for (int i = 0; i < NUM_SPLINES; ++i) hb[i] = (_Float16)B[i];
    *(half8*)(Phi + (size_t)token * KTOT + d * NUM_SPLINES) = hb;

    // Activation features: Phi[token][512 + d*4 + a] = {relu, tanh, sigmoid, x}
    float r  = xv > 0.0f ? xv : 0.0f;
    float th = tanhf(xv);
    float sg = 1.0f / (1.0f + expf(-xv));
    half4 ha;
    ha[0] = (_Float16)r;  ha[1] = (_Float16)th;
    ha[2] = (_Float16)sg; ha[3] = (_Float16)xv;
    *(half4*)(Phi + (size_t)token * KTOT + KSPL + d * 4) = ha;
}

// ---------------------------------------------------------------------------
// Kernel 2: pack weights Wm[o][k] = cp[o][k] (k<512) or 0.1*aw[o][k-512].
// blockIdx.x = o, threadIdx.x = k.
// ---------------------------------------------------------------------------
__global__ __launch_bounds__(KTOT)
void kan_weights(const float* __restrict__ cp,
                 const float* __restrict__ aw,
                 _Float16* __restrict__ Wm) {
    int o = blockIdx.x;
    int k = threadIdx.x;
    float v = (k < KSPL) ? cp[o * KSPL + k]
                         : 0.1f * aw[o * KACT + (k - KSPL)];
    Wm[o * KTOT + k] = (_Float16)v;
}

// ---------------------------------------------------------------------------
// Kernel 3: WMMA GEMM. D[o, token] = sum_k Wm[o,k] * Phi[token,k].
// One wave per 16x16 output tile: M = OUT_DIM (4 tiles), N = tokens (128 tiles).
// 512 waves total; K = 768 -> 24 wmma_f32_16x16x32_f16 per wave.
//
// A-fragment (16x32 f16): lane half h covers
//   K in {kb+16h .. kb+16h+15} (contiguous halves, matching B).
// B-fragment (32x16 f16): lanes 0-15 hold K=kb..kb+15 of column N=lane,
//   lanes 16-31 hold K=kb+16..kb+31 of column N=lane-16 (contiguous halves).
// C/D: VGPR j -> (M = j + 8h, N = lane&15); lanes map to consecutive tokens
//   so stores into the 'bon' layout are coalesced 64B segments.
// ---------------------------------------------------------------------------
__global__ __launch_bounds__(256)
void kan_gemm(const _Float16* __restrict__ Wm,
              const _Float16* __restrict__ Phi,
              float* __restrict__ out) {
    int wave  = (blockIdx.x * blockDim.x + threadIdx.x) >> 5;  // 0..511
    int lane  = threadIdx.x & 31;
    int ntile = wave >> 2;     // 0..127 over tokens
    int mtile = wave & 3;      // 0..3   over output channels
    int h     = lane >> 4;
    int l16   = lane & 15;

    int orow  = mtile * 16 + l16;   // A-matrix row (output channel)
    int token = ntile * 16 + l16;   // B-matrix column (token)

    const _Float16* wrow = Wm  + (size_t)orow  * KTOT;
    const _Float16* prow = Phi + (size_t)token * KTOT;

    float8 acc = {};
#pragma unroll 4
    for (int kb = 0; kb < KTOT; kb += 32) {
        half16 a  = *(const half16*)(wrow + kb + h * 16);
        half16 b  = *(const half16*)(prow + kb + h * 16);
        acc = __builtin_amdgcn_wmma_f32_16x16x32_f16(
                  /*neg_a=*/false, a, /*neg_b=*/false, b,
                  /*c_mod=*/(short)0, acc, /*reuse_a=*/false, /*reuse_b=*/false);
    }

    // token -> (batch, n); out layout is [b][o][n]
    int bidx = token >> 9;
    int n    = token & 511;
    float* obase = out + (size_t)bidx * (OUT_DIM * SEQ) + n;
#pragma unroll
    for (int j = 0; j < 8; ++j) {
        int o = mtile * 16 + h * 8 + j;
        obase[(size_t)o * SEQ] = acc[j];
    }
}

// ---------------------------------------------------------------------------
extern "C" void kernel_launch(void* const* d_in, const int* in_sizes, int n_in,
                              void* d_out, int out_size, void* d_ws, size_t ws_size,
                              hipStream_t stream) {
    (void)in_sizes; (void)n_in; (void)out_size; (void)ws_size;

    const float* x     = (const float*)d_in[0];  // (4, 512, 64)
    const float* cp    = (const float*)d_in[1];  // (64, 64, 8)
    const float* knots = (const float*)d_in[2];  // (64, 64, 12)
    const float* aw    = (const float*)d_in[3];  // (64, 64, 4)
    float*       out   = (float*)d_out;          // (4, 64, 512)

    // Workspace: Phi (2048*768 f16 = 3 MB) then Wm (64*768 f16 = 96 KB).
    _Float16* Phi = (_Float16*)d_ws;
    _Float16* Wm  = Phi + (size_t)NTOK * KTOT;

    kan_features<<<(NTOK * IN_DIM) / 256, 256, 0, stream>>>(x, knots, Phi);
    kan_weights <<<OUT_DIM, KTOT, 0, stream>>>(cp, aw, Wm);
    kan_gemm    <<<(512 * 32) / 256, 256, 0, stream>>>(Wm, Phi, out);
}
